// Seq2SeqGRU_33243046871395
// MI455X (gfx1250) — matrix operand order repaired and checked by reading, not compiled, not run on hardware
//
#include <hip/hip_runtime.h>
#include <hip/hip_bf16.h>
#include <cstdint>
#include <cstddef>

typedef __bf16 bf16_t;
typedef __attribute__((ext_vector_type(16))) __bf16 v16bf;
typedef __attribute__((ext_vector_type(8)))  float  v8f;
typedef __attribute__((ext_vector_type(4)))  float  v4f;

constexpr int CB = 32;      // batch
constexpr int CS = 48;      // encoder seq len
constexpr int CE = 256;     // embed dim
constexpr int CH = 1024;    // hidden
constexpr int CV = 32000;   // vocab (fr)
constexpr int CG = 32;      // generated length

// ---------------------------------------------------------------------------
// Weight packing: fp32 [K,N] row-major -> bf16 packed per lane for the WMMA B operand (same k order as A).
// Tile (kt,nt) of 32x16; within tile: element index = lane*16 + v2, where
// lane = khalf*16 + col  (khalf = K-half, col = N within tile),
// v2 in 0..15 gives K = kt*32 + khalf*16 + v2.
// ---------------------------------------------------------------------------
__global__ __launch_bounds__(256) void pack_w_k(const float* __restrict__ W,
                                                bf16_t* __restrict__ P,
                                                int K, int N) {
  size_t i = (size_t)blockIdx.x * blockDim.x + threadIdx.x;
  size_t total = (size_t)K * N;
  if (i >= total) return;
  int v2   = (int)(i & 15);
  int lane = (int)((i >> 4) & 31);
  size_t tile = i >> 9;
  int ktiles = K >> 5;
  int kt = (int)(tile % ktiles);
  int nt = (int)(tile / ktiles);
  int col   = lane & 15;
  int khalf = lane >> 4;
  int kk = kt * 32 + khalf * 16 + v2;
  int nn = nt * 16 + col;
  P[i] = (bf16_t)W[(size_t)kk * N + nn];
}

// ---------------------------------------------------------------------------
// WMMA GEMM: C[M,N](f32) = A[M,K](f32, lda) @ Bpacked[K,N](bf16) + bias[N]
// One wave per block; each wave computes a (MT*16) x (NT*16) = 32x64 tile.
// A converted fp32->bf16 in-register. Requires M%32==0, N%64==0, K%32==0.
// ---------------------------------------------------------------------------
#define MT 2
#define NT 4

__global__ __launch_bounds__(32) void wmma_gemm_k(const float* __restrict__ A, int lda,
                                                  const bf16_t* __restrict__ Bp,
                                                  const float* __restrict__ bias,
                                                  float* __restrict__ C, size_t ldc,
                                                  int K) {
  const int lane  = threadIdx.x & 31;
  const int half  = lane >> 4;
  const int mr    = lane & 15;
  const int m0    = blockIdx.y * (MT * 16);
  const int nt0   = blockIdx.x * NT;     // first 16-wide column tile
  const int ktile = K >> 5;

  v8f acc[MT][NT];
#pragma unroll
  for (int i = 0; i < MT; ++i)
#pragma unroll
    for (int j = 0; j < NT; ++j)
#pragma unroll
      for (int r = 0; r < 8; ++r) acc[i][j][r] = 0.f;

  const float* arow[MT];
#pragma unroll
  for (int i = 0; i < MT; ++i)
    arow[i] = A + (size_t)(m0 + i * 16 + mr) * lda + 16 * half;

  const bf16_t* bptr[NT];
#pragma unroll
  for (int j = 0; j < NT; ++j)
    bptr[j] = Bp + ((size_t)(nt0 + j) * ktile) * 512 + lane * 16;

  for (int kc = 0; kc < ktile; ++kc) {
    v16bf a[MT];
#pragma unroll
    for (int i = 0; i < MT; ++i) {
      const float* p = arow[i] + kc * 32;
      v4f f0 = *(const v4f*)(p);
      v4f f1 = *(const v4f*)(p + 4);
      v4f f2 = *(const v4f*)(p + 8);
      v4f f3 = *(const v4f*)(p + 12);
      v16bf av;
      av[0]  = (bf16_t)f0.x; av[1]  = (bf16_t)f0.y; av[2]  = (bf16_t)f0.z; av[3]  = (bf16_t)f0.w;
      av[4]  = (bf16_t)f1.x; av[5]  = (bf16_t)f1.y; av[6]  = (bf16_t)f1.z; av[7]  = (bf16_t)f1.w;
      av[8]  = (bf16_t)f2.x; av[9]  = (bf16_t)f2.y; av[10] = (bf16_t)f2.z; av[11] = (bf16_t)f2.w;
      av[12] = (bf16_t)f3.x; av[13] = (bf16_t)f3.y; av[14] = (bf16_t)f3.z; av[15] = (bf16_t)f3.w;
      a[i] = av;
    }
    v16bf b[NT];
#pragma unroll
    for (int j = 0; j < NT; ++j)
      b[j] = *(const v16bf*)(bptr[j] + (size_t)kc * 512);

#pragma unroll
    for (int i = 0; i < MT; ++i)
#pragma unroll
      for (int j = 0; j < NT; ++j)
        acc[i][j] = __builtin_amdgcn_wmma_f32_16x16x32_bf16(
            false, a[i], false, b[j], (short)0, acc[i][j], false, false);
  }

#pragma unroll
  for (int i = 0; i < MT; ++i) {
#pragma unroll
    for (int j = 0; j < NT; ++j) {
      int col  = (nt0 + j) * 16 + mr;
      float bv = bias ? bias[col] : 0.f;
#pragma unroll
      for (int r = 0; r < 8; ++r) {
        int row = m0 + i * 16 + 8 * half + r;
        C[(size_t)row * ldc + col] = acc[i][j][r] + bv;
      }
    }
  }
}

// ---------------------------------------------------------------------------
// Small elementwise / reduction kernels
// ---------------------------------------------------------------------------
__device__ __forceinline__ float sigm_f(float x) { return 1.f / (1.f + expf(-x)); }

__global__ __launch_bounds__(256) void embed_k(const int* __restrict__ ids,
                                               const float* __restrict__ tab,
                                               float* __restrict__ outp) {
  size_t idx = (size_t)blockIdx.x * 256 + threadIdx.x;
  if (idx >= (size_t)CB * CS * CE) return;
  size_t row = idx / CE, c = idx % CE;
  outp[idx] = tab[(size_t)ids[row] * CE + c];
}

__global__ __launch_bounds__(256) void fillz_k(float* __restrict__ p, int n) {
  int idx = blockIdx.x * 256 + threadIdx.x;
  if (idx < n) p[idx] = 0.f;
}

// GRU gate update, sequence form: GI rows indexed by (b*CS+t), GH rows by b.
__global__ __launch_bounds__(256) void gru_elem_k(const float* __restrict__ GI,
                                                  const float* __restrict__ GH,
                                                  float* __restrict__ h,
                                                  float* __restrict__ outp, int t) {
  int idx = blockIdx.x * 256 + threadIdx.x;
  if (idx >= CB * CH) return;
  int b = idx / CH, c = idx % CH;
  const float* gi = GI + (size_t)(b * CS + t) * (3 * CH);
  const float* gh = GH + (size_t)b * (3 * CH);
  float z = sigm_f(gi[c] + gh[c]);
  float r = sigm_f(gi[CH + c] + gh[CH + c]);
  float n = tanhf(gi[2 * CH + c] + r * gh[2 * CH + c]);
  float hv = z * h[idx] + (1.f - z) * n;
  h[idx] = hv;
  outp[(size_t)(b * CS + t) * CH + c] = hv;
}

// GRU step with h == 0 (decoder): gh = bias row only; h' = (1-z)*n
__global__ __launch_bounds__(256) void gru_zero_k(const float* __restrict__ GI,
                                                  const float* __restrict__ bh,
                                                  float* __restrict__ hout) {
  int idx = blockIdx.x * 256 + threadIdx.x;
  if (idx >= CB * CH) return;
  int b = idx / CH, c = idx % CH;
  const float* gi = GI + (size_t)b * (3 * CH);
  float z = sigm_f(gi[c] + bh[c]);
  float r = sigm_f(gi[CH + c] + bh[CH + c]);
  float n = tanhf(gi[2 * CH + c] + r * bh[2 * CH + c]);
  hout[idx] = (1.f - z) * n;
}

// score[b,s] = tanh(q[b]+proj[b,s]) . V + Vb ; one wave per (b,s)
__global__ __launch_bounds__(256) void attn_score_k(const float* __restrict__ q,
                                                    const float* __restrict__ proj,
                                                    const float* __restrict__ Vv,
                                                    const float* __restrict__ Vb,
                                                    float* __restrict__ score) {
  int w    = blockIdx.x * (blockDim.x / 32) + (threadIdx.x >> 5);
  int lane = threadIdx.x & 31;
  if (w >= CB * CS) return;
  int b = w / CS;
  const float* qp = q + (size_t)b * CH;
  const float* pp = proj + (size_t)w * CH;
  float sum = 0.f;
  for (int k = lane; k < CH; k += 32) sum += tanhf(qp[k] + pp[k]) * Vv[k];
#pragma unroll
  for (int m = 16; m > 0; m >>= 1) sum += __shfl_xor(sum, m, 32);
  if (lane == 0) score[w] = sum + Vb[0];
}

// softmax over S then ctx[b] = sum_s w[s]*enc_out[b,s,:]; one block per b
__global__ __launch_bounds__(256) void softmax_ctx_k(const float* __restrict__ score,
                                                     const float* __restrict__ enc_out,
                                                     float* __restrict__ ctx) {
  __shared__ float w[CS];
  __shared__ float sred;
  int b = blockIdx.x, tid = threadIdx.x;
  if (tid == 0) {
    float mx = -3.4e38f;
    for (int s = 0; s < CS; ++s) mx = fmaxf(mx, score[b * CS + s]);
    sred = mx;
  }
  __syncthreads();
  float mx = sred;
  if (tid < CS) w[tid] = expf(score[b * CS + tid] - mx);
  __syncthreads();
  if (tid == 0) {
    float sm = 0.f;
    for (int s = 0; s < CS; ++s) sm += w[s];
    sred = sm;
  }
  __syncthreads();
  float inv = 1.f / sred;
  for (int c = tid; c < CH; c += blockDim.x) {
    float acc = 0.f;
    for (int s = 0; s < CS; ++s) acc += w[s] * enc_out[(size_t)(b * CS + s) * CH + c];
    ctx[(size_t)b * CH + c] = acc * inv;
  }
}

// xt = concat(ctx, emb_fr[dec_in])
__global__ __launch_bounds__(256) void build_xt_k(const float* __restrict__ ctx,
                                                  const float* __restrict__ emb_fr,
                                                  const int* __restrict__ dec_in,
                                                  float* __restrict__ xt) {
  int idx = blockIdx.x * 256 + threadIdx.x;
  const int W = CH + CE;
  if (idx >= CB * W) return;
  int b = idx / W, c = idx % W;
  xt[idx] = (c < CH) ? ctx[(size_t)b * CH + c]
                     : emb_fr[(size_t)dec_in[b] * CE + (c - CH)];
}

// argmax over vocab (first-max tie rule), logits row at out[(b*CG+t)*CV]
__global__ __launch_bounds__(256) void argmax_k(const float* __restrict__ outp, int t,
                                                int* __restrict__ dec_in) {
  __shared__ float sv[256];
  __shared__ int   si[256];
  int b = blockIdx.x, tid = threadIdx.x;
  const float* row = outp + ((size_t)b * CG + t) * CV;
  float best = -3.4e38f;
  int   bi   = 0;
  for (int n = tid; n < CV; n += 256) {
    float v = row[n];
    if (v > best) { best = v; bi = n; }
  }
  sv[tid] = best; si[tid] = bi;
  __syncthreads();
  for (int s = 128; s > 0; s >>= 1) {
    if (tid < s) {
      if (sv[tid + s] > sv[tid] || (sv[tid + s] == sv[tid] && si[tid + s] < si[tid])) {
        sv[tid] = sv[tid + s]; si[tid] = si[tid + s];
      }
    }
    __syncthreads();
  }
  if (tid == 0) dec_in[b] = si[0];
}

// d_out[:,0,:] = one_hot(start_id)
__global__ __launch_bounds__(256) void onehot_k(float* __restrict__ outp,
                                                const int* __restrict__ start) {
  size_t idx = (size_t)blockIdx.x * 256 + threadIdx.x;
  if (idx >= (size_t)CB * CV) return;
  size_t b = idx / CV, n = idx % CV;
  outp[b * (size_t)CG * CV + n] = ((int)n == *start) ? 1.f : 0.f;
}

__global__ void seti_k(int* __restrict__ dec, const int* __restrict__ start) {
  if (threadIdx.x < CB) dec[threadIdx.x] = *start;
}

// ---------------------------------------------------------------------------
// Host orchestration
// ---------------------------------------------------------------------------
extern "C" void kernel_launch(void* const* d_in, const int* in_sizes, int n_in,
                              void* d_out, int out_size, void* d_ws, size_t ws_size,
                              hipStream_t stream) {
  (void)in_sizes; (void)n_in; (void)out_size; (void)ws_size;

  const int*   inputs   = (const int*)  d_in[0];
  const int*   start_id = (const int*)  d_in[2];
  const float* emb_en   = (const float*)d_in[3];
  const float* emb_fr   = (const float*)d_in[4];
  const float* enc_W0   = (const float*)d_in[5];
  const float* enc_U0   = (const float*)d_in[6];
  const float* enc_b0   = (const float*)d_in[7];
  const float* enc_W1   = (const float*)d_in[8];
  const float* enc_U1   = (const float*)d_in[9];
  const float* enc_b1   = (const float*)d_in[10];
  const float* dec_W0   = (const float*)d_in[11];
  /* d_in[12] = dec_U0 unused: decoder state is reset to zero each step */
  const float* dec_b0   = (const float*)d_in[13];
  const float* dec_W1   = (const float*)d_in[14];
  /* d_in[15] = dec_U1 unused */
  const float* dec_b1   = (const float*)d_in[16];
  const float* attn_W1  = (const float*)d_in[17];
  const float* attn_b1  = (const float*)d_in[18];
  const float* attn_W2  = (const float*)d_in[19];
  const float* attn_b2  = (const float*)d_in[20];
  const float* attn_V   = (const float*)d_in[21];
  const float* attn_Vb  = (const float*)d_in[22];
  const float* fc_W     = (const float*)d_in[23];
  const float* fc_b     = (const float*)d_in[24];
  float* outp = (float*)d_out;

  // ---- workspace layout (256B aligned sub-buffers) ----
  size_t off = 0;
  auto alloc = [&](size_t bytes) -> char* {
    off = (off + 255) & ~(size_t)255;
    char* p = (char*)d_ws + off;
    off += bytes;
    return p;
  };
  bf16_t* pW_e0 = (bf16_t*)alloc((size_t)CE * 3 * CH * 2);
  bf16_t* pU_e0 = (bf16_t*)alloc((size_t)CH * 3 * CH * 2);
  bf16_t* pW_e1 = (bf16_t*)alloc((size_t)CH * 3 * CH * 2);
  bf16_t* pU_e1 = (bf16_t*)alloc((size_t)CH * 3 * CH * 2);
  bf16_t* pW_d0 = (bf16_t*)alloc((size_t)(CH + CE) * 3 * CH * 2);
  bf16_t* pW_d1 = (bf16_t*)alloc((size_t)CH * 3 * CH * 2);
  bf16_t* pA1   = (bf16_t*)alloc((size_t)CH * CH * 2);
  bf16_t* pA2   = (bf16_t*)alloc((size_t)CH * CH * 2);
  bf16_t* pFC   = (bf16_t*)alloc((size_t)CH * CV * 2);

  float* X        = (float*)alloc((size_t)CB * CS * CE * 4);
  float* X1       = (float*)alloc((size_t)CB * CS * CH * 4);
  float* GI       = (float*)alloc((size_t)CB * CS * 3 * CH * 4);
  float* GH       = (float*)alloc((size_t)CB * 3 * CH * 4);
  float* h        = (float*)alloc((size_t)CB * CH * 4);   // encoder h -> decoder hidden
  float* h1       = (float*)alloc((size_t)CB * CH * 4);
  float* enc_out  = (float*)alloc((size_t)CB * CS * CH * 4);
  float* enc_proj = (float*)alloc((size_t)CB * CS * CH * 4);
  float* q        = (float*)alloc((size_t)CB * CH * 4);
  float* score    = (float*)alloc((size_t)CB * CS * 4);
  float* ctx      = (float*)alloc((size_t)CB * CH * 4);
  float* xt       = (float*)alloc((size_t)CB * (CH + CE) * 4);
  float* dGI      = (float*)alloc((size_t)CB * 3 * CH * 4);
  int*   dec_in   = (int*)  alloc((size_t)CB * 4);

  auto cdiv = [](long a, long b) { return (int)((a + b - 1) / b); };

  auto pack = [&](const float* W, bf16_t* P, int K, int N) {
    long n = (long)K * N;
    pack_w_k<<<cdiv(n, 256), 256, 0, stream>>>(W, P, K, N);
  };
  auto gemm = [&](const float* A, int lda, const bf16_t* Bp, const float* bias,
                  float* C, size_t ldc, int M, int N, int K) {
    dim3 g(N / 64, M / 32);
    wmma_gemm_k<<<g, 32, 0, stream>>>(A, lda, Bp, bias, C, ldc, K);
  };

  // ---- one-time weight packing (fp32 -> WMMA-packed bf16) ----
  pack(enc_W0, pW_e0, CE, 3 * CH);
  pack(enc_U0, pU_e0, CH, 3 * CH);
  pack(enc_W1, pW_e1, CH, 3 * CH);
  pack(enc_U1, pU_e1, CH, 3 * CH);
  pack(dec_W0, pW_d0, CH + CE, 3 * CH);
  pack(dec_W1, pW_d1, CH, 3 * CH);
  pack(attn_W1, pA1, CH, CH);
  pack(attn_W2, pA2, CH, CH);
  pack(fc_W,   pFC, CH, CV);

  // ---- encoder ----
  embed_k<<<cdiv((long)CB * CS * CE, 256), 256, 0, stream>>>(inputs, emb_en, X);

  // layer 0: GI = X @ W0 + b0[0] (batched over all timesteps)
  gemm(X, CE, pW_e0, enc_b0, GI, 3 * CH, CB * CS, 3 * CH, CE);
  fillz_k<<<cdiv(CB * CH, 256), 256, 0, stream>>>(h, CB * CH);
  for (int t = 0; t < CS; ++t) {
    gemm(h, CH, pU_e0, enc_b0 + 3 * CH, GH, 3 * CH, CB, 3 * CH, CH);
    gru_elem_k<<<cdiv(CB * CH, 256), 256, 0, stream>>>(GI, GH, h, X1, t);
  }

  // layer 1
  gemm(X1, CH, pW_e1, enc_b1, GI, 3 * CH, CB * CS, 3 * CH, CH);
  fillz_k<<<cdiv(CB * CH, 256), 256, 0, stream>>>(h, CB * CH);
  for (int t = 0; t < CS; ++t) {
    gemm(h, CH, pU_e1, enc_b1 + 3 * CH, GH, 3 * CH, CB, 3 * CH, CH);
    gru_elem_k<<<cdiv(CB * CH, 256), 256, 0, stream>>>(GI, GH, h, enc_out, t);
  }
  // h now holds enc_hidden -> decoder attention query state

  // enc_proj = enc_out @ attn_W2 + attn_b2
  gemm(enc_out, CH, pA2, attn_b2, enc_proj, CH, CB * CS, CH, CH);

  // ---- decoder ----
  onehot_k<<<cdiv((long)CB * CV, 256), 256, 0, stream>>>(outp, start_id);
  seti_k<<<1, 32, 0, stream>>>(dec_in, start_id);

  for (int t = 1; t < CG; ++t) {
    gemm(h, CH, pA1, attn_b1, q, CH, CB, CH, CH);
    attn_score_k<<<(CB * CS) / 8, 256, 0, stream>>>(q, enc_proj, attn_V, attn_Vb, score);
    softmax_ctx_k<<<CB, 256, 0, stream>>>(score, enc_out, ctx);
    build_xt_k<<<cdiv((long)CB * (CH + CE), 256), 256, 0, stream>>>(ctx, emb_fr, dec_in, xt);

    gemm(xt, CH + CE, pW_d0, dec_b0, dGI, 3 * CH, CB, 3 * CH, CH + CE);
    gru_zero_k<<<cdiv(CB * CH, 256), 256, 0, stream>>>(dGI, dec_b0 + 3 * CH, h1);
    gemm(h1, CH, pW_d1, dec_b1, dGI, 3 * CH, CB, 3 * CH, CH);
    gru_zero_k<<<cdiv(CB * CH, 256), 256, 0, stream>>>(dGI, dec_b1 + 3 * CH, h);

    // logits -> d_out[:, t, :]  (row stride CG*CV, base offset t*CV)
    gemm(h, CH, pFC, fc_b, outp + (size_t)t * CV, (size_t)CG * CV, CB, CV, CH);
    argmax_k<<<CB, 256, 0, stream>>>(outp, t, dec_in);
  }
}
